// MDM_27221502722124
// MI455X (gfx1250) — matrix operand order repaired and checked by reading, not compiled, not run on hardware
//
#include <hip/hip_runtime.h>
#include <hip/hip_bf16.h>
#include <stdint.h>

// Problem constants (match reference)
#define T_  4096
#define B_  4
#define D_  1024
#define H_  8
#define HD_ 128
#define KP  40      // padded K stride (bf16 elems) for LDS tiles: 80B rows keep uint4 alignment

typedef __attribute__((ext_vector_type(16))) __bf16 v16bf;
typedef __attribute__((ext_vector_type(8)))  float  v8f;
typedef __attribute__((ext_vector_type(4)))  int    v4i;

// ---- CDNA5 async global->LDS copy (ASYNCcnt-tracked), with sync fallback ----
#if defined(__has_builtin)
#  if __has_builtin(__builtin_amdgcn_global_load_async_to_lds_b128)
#    define HAVE_ASYNC_LDS 1
#  endif
#endif
#ifndef HAVE_ASYNC_LDS
#  define HAVE_ASYNC_LDS 0
#endif

__device__ __forceinline__ void cp16_async(const unsigned short* g, unsigned short* l) {
#if HAVE_ASYNC_LDS
  typedef __attribute__((address_space(1))) v4i GV;   // global int4
  typedef __attribute__((address_space(3))) v4i LV;   // LDS int4
  __builtin_amdgcn_global_load_async_to_lds_b128(
      (GV*)(const __attribute__((address_space(1))) void*)g,
      (LV*)(__attribute__((address_space(3))) void*)l,
      0, 0);
#else
  *(uint4*)l = *(const uint4*)g;
#endif
}

__device__ __forceinline__ void async_wait0() {
#if HAVE_ASYNC_LDS
#  if __has_builtin(__builtin_amdgcn_s_wait_asynccnt)
  __builtin_amdgcn_s_wait_asynccnt(0);
#  else
  asm volatile("s_wait_asynccnt 0x0" ::: "memory");
#  endif
#endif
}
__device__ __forceinline__ void async_wait4() {
#if HAVE_ASYNC_LDS
#  if __has_builtin(__builtin_amdgcn_s_wait_asynccnt)
  __builtin_amdgcn_s_wait_asynccnt(4);
#  else
  asm volatile("s_wait_asynccnt 0x4" ::: "memory");
#  endif
#endif
}

__device__ __forceinline__ unsigned short f2bf(float f) {
  union { float f; unsigned u; } v; v.f = f;
  unsigned r = v.u + 0x7FFFu + ((v.u >> 16) & 1u);  // round-to-nearest-even
  return (unsigned short)(r >> 16);
}

union AFrag { v16bf v; unsigned u[8]; };
union BFrag { v16bf v; uint4 q[2]; };
union LD16  { uint4 q[2]; unsigned short s[16]; };

// ---------------------------------------------------------------------------
// LayerNorm: one pass computes shared mean/rstd, emits BOTH gamma/beta variants
// as bf16 rows (query-branch xn and key/value-branch xtn).
// ---------------------------------------------------------------------------
__global__ __launch_bounds__(256) void ln_kernel(
    const float* __restrict__ x,
    const float* __restrict__ nw, const float* __restrict__ nb,
    const float* __restrict__ tw, const float* __restrict__ tb,
    unsigned short* __restrict__ xn, unsigned short* __restrict__ xtn)
{
  const int row = blockIdx.x;                 // row = b*T + t
  const int b = row / T_, t = row % T_;
  const float* src = x + ((size_t)t * B_ + b) * D_;  // x is (T,B,D)
  const int tid = threadIdx.x;

  float v[4]; float s = 0.f, ss = 0.f;
  #pragma unroll
  for (int j = 0; j < 4; ++j) {
    int e = tid + j * 256;
    v[j] = src[e];
    s += v[j]; ss += v[j] * v[j];
  }
  __shared__ float rs[256], rq[256];
  rs[tid] = s; rq[tid] = ss;
  __syncthreads();
  for (int off = 128; off > 0; off >>= 1) {
    if (tid < off) { rs[tid] += rs[tid + off]; rq[tid] += rq[tid + off]; }
    __syncthreads();
  }
  const float mean = rs[0] * (1.f / D_);
  const float var  = rq[0] * (1.f / D_) - mean * mean;
  const float rstd = rsqrtf(var + 1e-5f);

  unsigned short* dn = xn  + (size_t)row * D_;
  unsigned short* dt = xtn + (size_t)row * D_;
  #pragma unroll
  for (int j = 0; j < 4; ++j) {
    int e = tid + j * 256;
    float z = (v[j] - mean) * rstd;
    dn[e] = f2bf(z * nw[e] + nb[e]);
    dt[e] = f2bf(z * tw[e] + tb[e]);
  }
}

// f32 -> bf16 convert (weights)
__global__ void cvt_bf16(const float* __restrict__ s, unsigned short* __restrict__ d, int n) {
  int i = blockIdx.x * blockDim.x + threadIdx.x;
  if (i < n) d[i] = f2bf(s[i]);
}

// ---------------------------------------------------------------------------
// WMMA GEMM: C(MxN) = A(MxK,bf16) * Bt(NxK,bf16)^T + bias
// MODE 0: f32 out (K branch)   MODE 1: bf16 out (V branch)
// MODE 2: fused softmax over the 128-wide N-block (== one head), bf16 out (Q)
// Block = 256 thr (8 waves), 128x128 output tile, K stepped by 32.
// Double-buffered LDS tiles filled by async global->LDS copies.
// ---------------------------------------------------------------------------
template<int MODE>
__global__ __launch_bounds__(256) void gemm_wmma(
    const unsigned short* __restrict__ A,
    const unsigned short* __restrict__ Bt,
    const float* __restrict__ bias,
    float* __restrict__ outF,
    unsigned short* __restrict__ outH,
    int M, int N, int K)
{
  __shared__ union {
    struct { unsigned short As[2][128 * KP]; unsigned short Bs[2][128 * KP]; } t;
    float Cs[128 * 132];   // MODE 2 softmax staging (aliased; used after loop)
  } sm;

  const int tid = threadIdx.x;
  const int bM = blockIdx.y * 128;
  const int bN = blockIdx.x * 128;

  const int lrow  = tid >> 1;
  const int lhalf = (tid & 1) * 16;
  const unsigned short* gA = A  + (size_t)(bM + lrow) * K + lhalf;
  const unsigned short* gB = Bt + (size_t)(bN + lrow) * K + lhalf;

  const int lane = tid & 31;
  const int wave = tid >> 5;
  const int wm = (wave & 3) * 32;
  const int wn = (wave >> 2) * 64;
  const int lm  = lane & 15;
  const int ahk = (lane >> 4) * 8;    // A K-base per lane half (ISA 16-bit A layout)
  const int bhk = (lane >> 4) * 8;    // B K-base per lane half

  v8f acc[2][4] = {};
  const int NSTEP = K >> 5;

  // prologue: stage tile 0 into buffer 0 (4 async b128 per thread per tile)
  {
    cp16_async(gA,     &sm.t.As[0][lrow * KP + lhalf]);
    cp16_async(gA + 8, &sm.t.As[0][lrow * KP + lhalf + 8]);
    cp16_async(gB,     &sm.t.Bs[0][lrow * KP + lhalf]);
    cp16_async(gB + 8, &sm.t.Bs[0][lrow * KP + lhalf + 8]);
  }

  for (int it = 0; it < NSTEP; ++it) {
    const int cur = it & 1;
    if (it + 1 < NSTEP) {               // stage next tile into the other buffer
      const int k1 = (it + 1) << 5;
      cp16_async(gA + k1,     &sm.t.As[cur ^ 1][lrow * KP + lhalf]);
      cp16_async(gA + k1 + 8, &sm.t.As[cur ^ 1][lrow * KP + lhalf + 8]);
      cp16_async(gB + k1,     &sm.t.Bs[cur ^ 1][lrow * KP + lhalf]);
      cp16_async(gB + k1 + 8, &sm.t.Bs[cur ^ 1][lrow * KP + lhalf + 8]);
      async_wait4();                    // tile `it` landed; next tile in flight
    } else {
      async_wait0();
    }
    __syncthreads();

    AFrag af[2]; BFrag bf[4];
    #pragma unroll
    for (int tm = 0; tm < 2; ++tm) {
      const unsigned* rp = (const unsigned*)&sm.t.As[cur][(wm + tm * 16 + lm) * KP + ahk];
      af[tm].u[0]=rp[0]; af[tm].u[1]=rp[1]; af[tm].u[2]=rp[2];  af[tm].u[3]=rp[3];
      af[tm].u[4]=rp[8]; af[tm].u[5]=rp[9]; af[tm].u[6]=rp[10]; af[tm].u[7]=rp[11];
    }
    #pragma unroll
    for (int tn = 0; tn < 4; ++tn) {
      const uint4* rp = (const uint4*)&sm.t.Bs[cur][(wn + tn * 16 + lm) * KP + bhk];
      bf[tn].q[0] = rp[0]; bf[tn].q[1] = rp[2];
    }
    #pragma unroll
    for (int tm = 0; tm < 2; ++tm)
      #pragma unroll
      for (int tn = 0; tn < 4; ++tn)
        acc[tm][tn] = __builtin_amdgcn_wmma_f32_16x16x32_bf16(
            false, af[tm].v, false, bf[tn].v, (short)0, acc[tm][tn], false, false);
    __syncthreads();                    // everyone done reading buf[cur]
  }

  const int cn  = lane & 15;
  const int rmo = (lane >> 4) * 8;

  if (MODE == 2) {
    // Park the full 128x128 tile in LDS, softmax each row (one head), emit bf16.
    #pragma unroll
    for (int tm = 0; tm < 2; ++tm)
      for (int tn = 0; tn < 4; ++tn)
        #pragma unroll
        for (int r = 0; r < 8; ++r) {
          int rl = wm + tm * 16 + rmo + r;
          int cl = wn + tn * 16 + cn;
          sm.Cs[rl * 132 + cl] = acc[tm][tn][r] + bias[bN + cl];
        }
    __syncthreads();
    if (tid < 128) {
      const float* rowp = &sm.Cs[tid * 132];
      float mx = -3.4e38f;
      for (int j = 0; j < 128; ++j) mx = fmaxf(mx, rowp[j]);
      float s = 0.f;
      for (int j = 0; j < 128; ++j) s += __expf(rowp[j] - mx);
      float inv = 1.f / s;
      unsigned short* dst = outH + (size_t)(bM + tid) * N + bN;
      for (int j = 0; j < 128; ++j) dst[j] = f2bf(__expf(rowp[j] - mx) * inv);
    }
  } else {
    #pragma unroll
    for (int tm = 0; tm < 2; ++tm)
      for (int tn = 0; tn < 4; ++tn)
        #pragma unroll
        for (int r = 0; r < 8; ++r) {
          int row = bM + wm + tm * 16 + rmo + r;
          int col = bN + wn + tn * 16 + cn;
          float val = acc[tm][tn][r] + bias[col];
          if (MODE == 0) outF[(size_t)row * N + col] = val;
          else           outH[(size_t)row * N + col] = f2bf(val);
        }
  }
}

// ---------------------------------------------------------------------------
// Per-column (over time) online max/sum-exp for K softmax.  Grid (D/64, B).
// ---------------------------------------------------------------------------
__global__ __launch_bounds__(256) void kcolstats(const float* __restrict__ k,
                                                 float* __restrict__ kmax,
                                                 float* __restrict__ ksum)
{
  const int b = blockIdx.y;
  const int d0 = blockIdx.x * 64;
  const int dl = threadIdx.x & 63;
  const int tseg = threadIdx.x >> 6;
  const float* base = k + (size_t)b * T_ * D_ + d0 + dl;
  float m = -3.4e38f, s = 0.f;
  for (int t = tseg; t < T_; t += 4) {
    float val = base[(size_t)t * D_];
    float nm = fmaxf(m, val);
    s = s * __expf(m - nm) + __expf(val - nm);
    m = nm;
  }
  __shared__ float smx[4][64], ssm[4][64];
  smx[tseg][dl] = m; ssm[tseg][dl] = s;
  __syncthreads();
  if (tseg == 0) {
    #pragma unroll
    for (int i = 1; i < 4; ++i) {
      float m2 = smx[i][dl], s2 = ssm[i][dl];
      float nm = fmaxf(m, m2);
      s = s * __expf(m - nm) + s2 * __expf(m2 - nm);
      m = nm;
    }
    kmax[b * D_ + d0 + dl] = m;
    ksum[b * D_ + d0 + dl] = s;
  }
}

// Normalize K over time -> bf16 softk
__global__ void knorm(const float* __restrict__ k, const float* __restrict__ kmax,
                      const float* __restrict__ ksum, unsigned short* __restrict__ sk)
{
  const size_t total = (size_t)B_ * T_ * D_;
  for (size_t i = (size_t)blockIdx.x * blockDim.x + threadIdx.x; i < total;
       i += (size_t)gridDim.x * blockDim.x) {
    int d  = (int)(i % D_);
    int bt = (int)(i / D_);
    int b  = bt / T_;
    int col = b * D_ + d;
    sk[i] = f2bf(__expf(k[i] - kmax[col]) * (1.f / ksum[col]));
  }
}

// ---------------------------------------------------------------------------
// att[b,h] (128x128) = softk[b,:,h]^T (128xT) @ v[b,:,h] (Tx128).
// Grid (H, B); both operands staged K-major in LDS via transpose-on-store
// (transpose => must go through VGPRs, async copy not applicable here).
// ---------------------------------------------------------------------------
__global__ __launch_bounds__(256) void att_wmma(
    const unsigned short* __restrict__ sk,
    const unsigned short* __restrict__ vb,
    unsigned short* __restrict__ att)
{
  __shared__ unsigned short As[128 * KP];
  __shared__ unsigned short Bs[128 * KP];
  const int h = blockIdx.x, b = blockIdx.y;
  const int tid = threadIdx.x;
  const int tt  = tid >> 3;             // 0..31 : t within tile
  const int seg = (tid & 7) * 16;       // 16-wide column segment
  const unsigned short* gK = sk + ((size_t)b * T_) * D_ + h * HD_ + seg;
  const unsigned short* gV = vb + ((size_t)b * T_) * D_ + h * HD_ + seg;

  const int lane = tid & 31, wave = tid >> 5;
  const int wm = (wave & 3) * 32, wn = (wave >> 2) * 64;
  const int lm  = lane & 15;
  const int ahk = (lane >> 4) * 8, bhk = (lane >> 4) * 8;

  v8f acc[2][4] = {};

  for (int t0 = 0; t0 < T_; t0 += 32) {
    LD16 uk, uv;
    const size_t ro = (size_t)(t0 + tt) * D_;
    uk.q[0] = *(const uint4*)(gK + ro);
    uk.q[1] = *(const uint4*)(gK + ro + 8);
    uv.q[0] = *(const uint4*)(gV + ro);
    uv.q[1] = *(const uint4*)(gV + ro + 8);
    #pragma unroll
    for (int j = 0; j < 16; ++j) {      // transpose-on-store: tiles become [d][t]
      As[(seg + j) * KP + tt] = uk.s[j];
      Bs[(seg + j) * KP + tt] = uv.s[j];
    }
    __syncthreads();

    AFrag af[2]; BFrag bf[4];
    #pragma unroll
    for (int tm = 0; tm < 2; ++tm) {
      const unsigned* rp = (const unsigned*)&As[(wm + tm * 16 + lm) * KP + ahk];
      af[tm].u[0]=rp[0]; af[tm].u[1]=rp[1]; af[tm].u[2]=rp[2];  af[tm].u[3]=rp[3];
      af[tm].u[4]=rp[8]; af[tm].u[5]=rp[9]; af[tm].u[6]=rp[10]; af[tm].u[7]=rp[11];
    }
    #pragma unroll
    for (int tn = 0; tn < 4; ++tn) {
      const uint4* rp = (const uint4*)&Bs[(wn + tn * 16 + lm) * KP + bhk];
      bf[tn].q[0] = rp[0]; bf[tn].q[1] = rp[2];
    }
    #pragma unroll
    for (int tm = 0; tm < 2; ++tm)
      #pragma unroll
      for (int tn = 0; tn < 4; ++tn)
        acc[tm][tn] = __builtin_amdgcn_wmma_f32_16x16x32_bf16(
            false, af[tm].v, false, bf[tn].v, (short)0, acc[tm][tn], false, false);
    __syncthreads();
  }

  const int cn = lane & 15, rmo = (lane >> 4) * 8;
  unsigned short* base = att + (size_t)(b * H_ + h) * HD_ * HD_;
  #pragma unroll
  for (int tm = 0; tm < 2; ++tm)
    for (int tn = 0; tn < 4; ++tn)
      #pragma unroll
      for (int r = 0; r < 8; ++r) {
        int row = wm + tm * 16 + rmo + r;
        int col = wn + tn * 16 + cn;
        base[row * HD_ + col] = f2bf(acc[tm][tn][r]);
      }
}

// ---------------------------------------------------------------------------
// y[b,:,h] (Tx128) = qs[b,:,h] (Tx128) @ att[b,h] (128x128).  Grid (T/128,H,B)
// att staged transposed ([l][d]) once per block; qs tiles staged per K step
// via async global->LDS copies.
// ---------------------------------------------------------------------------
__global__ __launch_bounds__(256) void y_wmma(
    const unsigned short* __restrict__ qs,
    const unsigned short* __restrict__ att,
    float* __restrict__ out)
{
  __shared__ unsigned short As[128 * KP];
  __shared__ unsigned short Bs[128 * 136];
  const int mt = blockIdx.x, h = blockIdx.y, b = blockIdx.z;
  const int tid = threadIdx.x;

  {  // stage att^T: Bs[l][d]
    const unsigned short* ga = att + (size_t)(b * H_ + h) * HD_ * HD_;
    const int dr = tid >> 1, cs = (tid & 1) * 64;
    union { uint4 q[8]; unsigned short s[64]; } u;
    const uint4* gp = (const uint4*)(ga + dr * HD_ + cs);
    #pragma unroll
    for (int j = 0; j < 8; ++j) u.q[j] = gp[j];
    #pragma unroll
    for (int j = 0; j < 64; ++j) Bs[(cs + j) * 136 + dr] = u.s[j];
  }

  const int lrow = tid >> 1, lhalf = (tid & 1) * 16;
  const unsigned short* gq =
      qs + ((size_t)b * T_ + mt * 128 + lrow) * D_ + h * HD_ + lhalf;

  const int lane = tid & 31, wave = tid >> 5;
  const int wm = (wave & 3) * 32, wn = (wave >> 2) * 64;
  const int lm  = lane & 15;
  const int ahk = (lane >> 4) * 8, bhk = (lane >> 4) * 8;

  v8f acc[2][4] = {};
  for (int k0 = 0; k0 < HD_; k0 += 32) {
    cp16_async(gq + k0,     &As[lrow * KP + lhalf]);
    cp16_async(gq + k0 + 8, &As[lrow * KP + lhalf + 8]);
    async_wait0();
    __syncthreads();

    AFrag af[2]; BFrag bf[4];
    #pragma unroll
    for (int tm = 0; tm < 2; ++tm) {
      const unsigned* rp = (const unsigned*)&As[(wm + tm * 16 + lm) * KP + ahk];
      af[tm].u[0]=rp[0]; af[tm].u[1]=rp[1]; af[tm].u[2]=rp[2];  af[tm].u[3]=rp[3];
      af[tm].u[4]=rp[8]; af[tm].u[5]=rp[9]; af[tm].u[6]=rp[10]; af[tm].u[7]=rp[11];
    }
    #pragma unroll
    for (int tn = 0; tn < 4; ++tn) {
      const uint4* rp = (const uint4*)&Bs[(wn + tn * 16 + lm) * 136 + k0 + bhk];
      bf[tn].q[0] = rp[0]; bf[tn].q[1] = rp[2];
    }
    #pragma unroll
    for (int tm = 0; tm < 2; ++tm)
      #pragma unroll
      for (int tn = 0; tn < 4; ++tn)
        acc[tm][tn] = __builtin_amdgcn_wmma_f32_16x16x32_bf16(
            false, af[tm].v, false, bf[tn].v, (short)0, acc[tm][tn], false, false);
    __syncthreads();
  }

  const int cn = lane & 15, rmo = (lane >> 4) * 8;
  #pragma unroll
  for (int tm = 0; tm < 2; ++tm)
    for (int tn = 0; tn < 4; ++tn)
      #pragma unroll
      for (int r = 0; r < 8; ++r) {
        int row = mt * 128 + wm + tm * 16 + rmo + r;
        int col = wn + tn * 16 + cn;
        out[((size_t)b * T_ + row) * D_ + h * HD_ + col] = acc[tm][tn][r];
      }
}

// ---------------------------------------------------------------------------
extern "C" void kernel_launch(void* const* d_in, const int* in_sizes, int n_in,
                              void* d_out, int out_size, void* d_ws, size_t ws_size,
                              hipStream_t stream)
{
  const float* x  = (const float*)d_in[0];
  const float* nw = (const float*)d_in[1];
  const float* nb = (const float*)d_in[2];
  const float* tw = (const float*)d_in[3];
  const float* tb = (const float*)d_in[4];
  const float* Wq = (const float*)d_in[5];
  const float* bq = (const float*)d_in[6];
  const float* Wk = (const float*)d_in[7];
  const float* bk = (const float*)d_in[8];
  const float* Wv = (const float*)d_in[9];
  const float* bv = (const float*)d_in[10];
  float* out = (float*)d_out;
  char* ws = (char*)d_ws;

  const size_t ROWS   = (size_t)B_ * T_;                       // 16384
  const size_t SZ_X16 = ROWS * D_ * sizeof(unsigned short);    // 32 MB
  const size_t SZ_W16 = (size_t)D_ * D_ * sizeof(unsigned short);
  const size_t SZ_KF  = ROWS * D_ * sizeof(float);             // 64 MB

  size_t off = 0;
  unsigned short* xn   = (unsigned short*)(ws + off); off += SZ_X16;
  unsigned short* xtn  = (unsigned short*)(ws + off); off += SZ_X16;
  unsigned short* wqb  = (unsigned short*)(ws + off); off += SZ_W16;
  unsigned short* wkb  = (unsigned short*)(ws + off); off += SZ_W16;
  unsigned short* wvb  = (unsigned short*)(ws + off); off += SZ_W16;
  float*          kf   = (float*)(ws + off);          off += SZ_KF;
  unsigned short* vb16 = (unsigned short*)(ws + off); off += SZ_X16;
  unsigned short* qs16 = (unsigned short*)(ws + off); off += SZ_X16;
  unsigned short* sk16 = (unsigned short*)(ws + off); off += SZ_X16;
  float*          kmax = (float*)(ws + off);          off += (size_t)B_ * D_ * sizeof(float);
  float*          ksum = (float*)(ws + off);          off += (size_t)B_ * D_ * sizeof(float);
  unsigned short* at16 = (unsigned short*)(ws + off); off += (size_t)B_ * H_ * HD_ * HD_ * sizeof(unsigned short);

  // 1) fused double-LayerNorm -> bf16 activations
  ln_kernel<<<(unsigned)ROWS, 256, 0, stream>>>(x, nw, nb, tw, tb, xn, xtn);

  // 2) weights -> bf16 (row-major NxK is exactly the WMMA B-operand layout)
  const int nW = D_ * D_;
  cvt_bf16<<<nW / 256, 256, 0, stream>>>(Wq, wqb, nW);
  cvt_bf16<<<nW / 256, 256, 0, stream>>>(Wk, wkb, nW);
  cvt_bf16<<<nW / 256, 256, 0, stream>>>(Wv, wvb, nW);

  // 3) QKV GEMMs (Q fuses the head-dim softmax in its epilogue)
  dim3 gg(D_ / 128, (unsigned)(ROWS / 128));
  gemm_wmma<2><<<gg, 256, 0, stream>>>(xn,  wqb, bq, nullptr, qs16, (int)ROWS, D_, D_);
  gemm_wmma<0><<<gg, 256, 0, stream>>>(xtn, wkb, bk, kf,      nullptr, (int)ROWS, D_, D_);
  gemm_wmma<1><<<gg, 256, 0, stream>>>(xtn, wvb, bv, nullptr, vb16,  (int)ROWS, D_, D_);

  // 4) time-softmax for K: column stats then normalize
  kcolstats<<<dim3(D_ / 64, B_), 256, 0, stream>>>(kf, kmax, ksum);
  knorm<<<4096, 256, 0, stream>>>(kf, kmax, ksum, sk16);

  // 5) att = softk^T @ v per (b,h), then y = qsoft @ att -> d_out
  att_wmma<<<dim3(H_, B_), 256, 0, stream>>>(sk16, vb16, at16);
  y_wmma<<<dim3(T_ / 128, H_, B_), 256, 0, stream>>>(qs16, at16, out);
}
